// Transformer_54262616818356
// MI455X (gfx1250) — matrix operand order repaired and checked by reading, not compiled, not run on hardware
//
#include <hip/hip_runtime.h>
#include <hip/hip_bf16.h>
#include <math.h>
#include <stdint.h>

// ---------------- problem constants ----------------
#define BB 4
#define SS 1024
#define DD 512
#define HH 8
#define DHD 64
#define LL 6
#define FFD 2048
#define VV 16000
#define MM (BB * SS)        // 4096 rows of activations

// ---------------- WMMA types ----------------
typedef __attribute__((ext_vector_type(16))) __bf16 v16bf;
typedef __attribute__((ext_vector_type(8)))  float  v8f;

union FragU { unsigned u[8]; v16bf v; };

// f32 pair -> packed bf16 dword via single v_perm_b32 (truncate-to-bf16)
__device__ __forceinline__ unsigned pk_bf16(float lo, float hi) {
    return __builtin_amdgcn_perm(__float_as_uint(hi), __float_as_uint(lo), 0x07060302u);
}

__device__ __forceinline__ v8f wmma_bf16(const FragU& a, const FragU& b, v8f c) {
    return __builtin_amdgcn_wmma_f32_16x16x32_bf16(false, a.v, false, b.v,
                                                   (short)0, c, false, false);
}

// LDS byte offset of a __shared__ object (flat address low 32 bits == LDS offset)
__device__ __forceinline__ unsigned lds_off(const void* p) {
    return (unsigned)(uintptr_t)p;
}

// async global -> LDS, 16 bytes per lane, GVS mode (SGPR base + u32 lane offset)
__device__ __forceinline__ void async_b128(unsigned lds, unsigned goff, const float* base) {
    asm volatile("global_load_async_to_lds_b128 %0, %1, %2"
                 :: "v"(lds), "v"(goff), "s"(base) : "memory");
}

__device__ __forceinline__ void wait_async0() {
#if __has_builtin(__builtin_amdgcn_s_wait_asynccnt)
    __builtin_amdgcn_s_wait_asynccnt(0);
#else
    asm volatile("s_wait_asynccnt 0x0" ::: "memory");
#endif
}

// =====================================================================
// Embedding + sinusoidal positional encoding
// =====================================================================
__global__ __launch_bounds__(256) void embed_kernel(const int* __restrict__ tok,
                                                    const float* __restrict__ emb,
                                                    float* __restrict__ out,
                                                    int total) {
    int i = blockIdx.x * 256 + threadIdx.x;
    if (i >= total) return;
    int d  = i & (DD - 1);
    int bs = i >> 9;                 // DD = 512 = 2^9
    int s  = bs & (SS - 1);
    int t  = tok[bs];
    float ang = (float)s * __expf(-(float)(d & ~1) * (9.2103403719761836f / (float)DD));
    float pe  = (d & 1) ? __cosf(ang) : __sinf(ang);
    out[i] = emb[(size_t)t * DD + d] + pe;
}

// =====================================================================
// GEMM: C[M,N] = act(A[M,K] @ W[K,N] + bias[N]), fp32 in/out,
// bf16 WMMA with fp32 accumulation.
// Block tile 128(M) x 128(N), K-step 32, 8 waves; each wave owns
// 2 m-tiles x 4 n-tiles = 8 WMMA accumulators. All fragments are
// preloaded per K-step so the 8 WMMAs issue as one hazard-free block.
// Tiles staged raw-f32 into double-buffered LDS via async B128 loads
// (ASYNCcnt path); bf16 conversion via v_perm at fragment build.
// =====================================================================
#define GBM 128
#define GBN 128
#define APD 36     // A row stride in floats (144B, 16B aligned)
#define BPD 132    // B row stride in floats (528B, 16B aligned)
__global__ __launch_bounds__(256) void gemm_bf16(const float* __restrict__ A,
                                                 const float* __restrict__ W,
                                                 const float* __restrict__ bias,
                                                 float* __restrict__ C,
                                                 int M, int N, int K, int relu) {
    __shared__ float ldsA[2][GBM][APD];   // raw f32 A tile, double buffered
    __shared__ float ldsB[2][32][BPD];    // raw f32 B tile, double buffered

    int tid  = threadIdx.x;
    int wave = tid >> 5;
    int lane = tid & 31;
    int half = lane >> 4;
    int lm   = lane & 15;

    int m0 = blockIdx.y * GBM;
    int n0 = blockIdx.x * GBN;

    int mg    = wave & 3;     // m-tile group: owns m-tiles 2*mg, 2*mg+1
    int nhalf = wave >> 2;    // n-tile group 0/1 (4 tiles each)

    v8f acc[2][4];
    #pragma unroll
    for (int mi = 0; mi < 2; ++mi)
        #pragma unroll
        for (int j = 0; j < 4; ++j) acc[mi][j] = (v8f){};

    // stage tile (k0): A 128x32 f32 (1024 quads), B 32x128 f32 (1024 quads)
    auto stage = [&](int buf, int k0) {
        #pragma unroll
        for (int i = 0; i < 4; ++i) {
            int idx = tid + i * 256;          // 0..1023
            int kq  = idx & 7;                // quad within 32-wide K slice
            int m   = idx >> 3;
            unsigned l = lds_off(&ldsA[buf][m][4 * kq]);
            unsigned g = (unsigned)(((m0 + m) * K + k0 + 4 * kq) * 4);
            async_b128(l, g, A);
        }
        #pragma unroll
        for (int i = 0; i < 4; ++i) {
            int idx = tid + i * 256;          // 0..1023
            int nq  = idx & 31;
            int k   = idx >> 5;
            unsigned l = lds_off(&ldsB[buf][k][4 * nq]);
            unsigned g = (unsigned)(((k0 + k) * N + n0 + 4 * nq) * 4);
            async_b128(l, g, W);
        }
    };

    stage(0, 0);
    int nk = K >> 5;
    for (int t = 0; t < nk; ++t) {
        wait_async0();           // own async writes done
        __syncthreads();         // everyone's writes done; prev reads done
        if (t + 1 < nk) stage((t + 1) & 1, (t + 1) * 32);
        int buf = t & 1;

        // A fragments (16x32 bf16, contiguous k = 16*half + s); pairs contiguous -> b64 reads
        FragU af[2];
        #pragma unroll
        for (int mi = 0; mi < 2; ++mi)
            #pragma unroll
            for (int v = 0; v < 8; ++v) {
                int kp = v + 8 * half;
                const float* p = &ldsA[buf][(2 * mg + mi) * 16 + lm][2 * kp];
                af[mi].u[v] = pk_bf16(p[0], p[1]);
            }
        // all 4 B fragments into distinct registers (pairs one row apart -> ds_load_2addr)
        FragU bfr[4];
        #pragma unroll
        for (int j = 0; j < 4; ++j)
            #pragma unroll
            for (int v = 0; v < 8; ++v) {
                int k = 2 * v + 16 * half;
                const float* p = &ldsB[buf][k][nhalf * 64 + j * 16 + lm];
                bfr[j].u[v] = pk_bf16(p[0], p[BPD]);
            }
        // 8 independent WMMAs back-to-back (hazard NOPs amortized once per K-step)
        #pragma unroll
        for (int j = 0; j < 4; ++j)
            #pragma unroll
            for (int mi = 0; mi < 2; ++mi)
                acc[mi][j] = wmma_bf16(af[mi], bfr[j], acc[mi][j]);
    }

    // epilogue: C layout -> lane holds col lm, rows r + 8*half
    #pragma unroll
    for (int j = 0; j < 4; ++j) {
        int col = n0 + nhalf * 64 + j * 16 + lm;
        float bv = bias[col];
        #pragma unroll
        for (int mi = 0; mi < 2; ++mi) {
            #pragma unroll
            for (int r = 0; r < 8; ++r) {
                int row = m0 + (2 * mg + mi) * 16 + r + 8 * half;
                float vres = acc[mi][j][r] + bv;
                if (relu) vres = fmaxf(vres, 0.0f);
                C[(size_t)row * N + col] = vres;
            }
        }
    }
}

// =====================================================================
// Fused attention for one (b, h, 32-query block):
//   scores = Q Kt / 8 -> mask -> softmax -> O = P V
// scores (32 x 1024 fp32) live in LDS (~132 KB; CDNA5 WGP has 320 KB).
// =====================================================================
#define QB 32
#define SCP (SS + 8)   // padded row stride
__global__ __launch_bounds__(256) void attn_kernel(const float* __restrict__ Qb,
                                                   const float* __restrict__ Kb,
                                                   const float* __restrict__ Vb,
                                                   float* __restrict__ Ob,
                                                   const int* __restrict__ tok,
                                                   int causal) {
    __shared__ float sc[QB][SCP];
    __shared__ float red[QB * 8];

    int b  = blockIdx.z;
    int h  = blockIdx.y;
    int q0 = blockIdx.x * QB;

    int tid  = threadIdx.x;
    int wave = tid >> 5;
    int lane = tid & 31;
    int half = lane >> 4;
    int lm   = lane & 15;

    int mt = wave & 1;   // query m-tile 0/1

    // ---- load Q fragments for this wave's m-tile (K dim = 64 -> 2 frags) ----
    FragU qa0, qa1;
    #pragma unroll
    for (int v = 0; v < 8; ++v) {
        int p = v + 8 * half;
        const float* p0 = Qb + (size_t)(b * SS + q0 + mt * 16 + lm) * DD + h * DHD + 2 * p;
        qa0.u[v] = pk_bf16(p0[0], p0[1]);
        qa1.u[v] = pk_bf16(p0[32], p0[33]);
    }

    // ---- scores: each wave covers 16 key tiles ----
    for (int nt = (wave >> 1); nt < SS / 16; nt += 4) {
        FragU kf[2];
        #pragma unroll
        for (int f = 0; f < 2; ++f)
            #pragma unroll
            for (int v = 0; v < 8; ++v) {
                int d = f * 32 + 2 * v + 16 * half;
                const float* p0 = Kb + (size_t)(b * SS + nt * 16 + lm) * DD + h * DHD + d;
                kf[f].u[v] = pk_bf16(p0[0], p0[1]);
            }
        v8f c = {};
        c = wmma_bf16(qa0, kf[0], c);
        c = wmma_bf16(qa1, kf[1], c);
        #pragma unroll
        for (int r = 0; r < 8; ++r)
            sc[mt * 16 + r + 8 * half][nt * 16 + lm] = c[r] * 0.125f;
    }
    __syncthreads();

    // ---- mask + softmax: 8 threads per row ----
    int row = tid >> 3;
    int t8  = tid & 7;
    int qg  = q0 + row;
    const int* ktok = tok + b * SS;

    float mx = -3.0e38f;
    for (int i = 0; i < SS / 8; ++i) {
        int col = t8 + 8 * i;
        float s = sc[row][col];
        bool keep = (ktok[col] != 0) && (!causal || (qg >= col));
        s = keep ? s : -1.0e9f;
        sc[row][col] = s;
        mx = fmaxf(mx, s);
    }
    red[row * 8 + t8] = mx;
    __syncthreads();
    float rm = red[row * 8];
    #pragma unroll
    for (int i = 1; i < 8; ++i) rm = fmaxf(rm, red[row * 8 + i]);

    float sm = 0.0f;
    for (int i = 0; i < SS / 8; ++i) {
        int col = t8 + 8 * i;
        float e = __expf(sc[row][col] - rm);
        sc[row][col] = e;
        sm += e;
    }
    __syncthreads();               // all reads of red (maxes) complete
    red[row * 8 + t8] = sm;
    __syncthreads();
    float rs = 0.0f;
    #pragma unroll
    for (int i = 0; i < 8; ++i) rs += red[row * 8 + i];
    float inv = 1.0f / rs;
    for (int i = 0; i < SS / 8; ++i) {
        int col = t8 + 8 * i;
        sc[row][col] *= inv;
    }
    __syncthreads();

    // ---- O = P V : wave owns (mt, nt2) 16x16 tile of the 32x64 output ----
    // unrolled by 2 k-steps with independent fragment sets
    int nt2 = wave >> 1;           // 0..3
    v8f acc = {};
    for (int kk = 0; kk < SS; kk += 64) {
        FragU pa[2], vf[2];
        #pragma unroll
        for (int u = 0; u < 2; ++u) {
            int kb = kk + 32 * u;
            #pragma unroll
            for (int v = 0; v < 8; ++v) {
                int p = v + 8 * half;
                const float* sp = &sc[mt * 16 + lm][kb + 2 * p];
                pa[u].u[v] = pk_bf16(sp[0], sp[1]);
                int kr = kb + 2 * v + 16 * half;
                const float* vp = Vb + (size_t)(b * SS + kr) * DD + h * DHD + nt2 * 16 + lm;
                vf[u].u[v] = pk_bf16(vp[0], vp[DD]);
            }
        }
        acc = wmma_bf16(pa[0], vf[0], acc);
        acc = wmma_bf16(pa[1], vf[1], acc);
    }
    #pragma unroll
    for (int r = 0; r < 8; ++r)
        Ob[(size_t)(b * SS + q0 + mt * 16 + r + 8 * half) * DD + h * DHD + nt2 * 16 + lm] = acc[r];
}

// =====================================================================
// out = LayerNorm(x + dlt) * g + b  — one row (D=512) per 32-lane wave
// =====================================================================
__global__ __launch_bounds__(256) void add_ln_kernel(const float* __restrict__ x,
                                                     const float* __restrict__ dlt,
                                                     const float* __restrict__ g,
                                                     const float* __restrict__ bta,
                                                     float* __restrict__ out) {
    int wave = threadIdx.x >> 5;
    int lane = threadIdx.x & 31;
    int row  = blockIdx.x * 8 + wave;

    const float* xr = x   + (size_t)row * DD;
    const float* dr = dlt + (size_t)row * DD;
    float v[16];
    float s = 0.0f;
    #pragma unroll
    for (int i = 0; i < 16; ++i) {
        int d = lane + 32 * i;
        v[i] = xr[d] + dr[d];
        s += v[i];
    }
    #pragma unroll
    for (int o = 16; o >= 1; o >>= 1) s += __shfl_xor(s, o, 32);
    float mean = s * (1.0f / DD);
    float var = 0.0f;
    #pragma unroll
    for (int i = 0; i < 16; ++i) { float t = v[i] - mean; var += t * t; }
    #pragma unroll
    for (int o = 16; o >= 1; o >>= 1) var += __shfl_xor(var, o, 32);
    var *= (1.0f / DD);
    float inv = rsqrtf(var + 1e-5f);
    float* orow = out + (size_t)row * DD;
    #pragma unroll
    for (int i = 0; i < 16; ++i) {
        int d = lane + 32 * i;
        orow[d] = (v[i] - mean) * inv * g[d] + bta[d];
    }
}

// =====================================================================
// Host orchestration
// =====================================================================
extern "C" void kernel_launch(void* const* d_in, const int* in_sizes, int n_in,
                              void* d_out, int out_size, void* d_ws, size_t ws_size,
                              hipStream_t stream) {
    const int*   src      = (const int*)  d_in[0];
    const int*   trg      = (const int*)  d_in[1];
    const float* enc_emb  = (const float*)d_in[2];
    const float* dec_emb  = (const float*)d_in[3];
    const float* e_attn_W = (const float*)d_in[4];
    const float* e_attn_b = (const float*)d_in[5];
    const float* e_ln_g   = (const float*)d_in[6];
    const float* e_ln_b   = (const float*)d_in[7];
    const float* e_W1     = (const float*)d_in[8];
    const float* e_b1     = (const float*)d_in[9];
    const float* e_W2     = (const float*)d_in[10];
    const float* e_b2     = (const float*)d_in[11];
    const float* d_self_W = (const float*)d_in[12];
    const float* d_self_b = (const float*)d_in[13];
    const float* d_crs_W  = (const float*)d_in[14];
    const float* d_crs_b  = (const float*)d_in[15];
    const float* d_ln_g   = (const float*)d_in[16];
    const float* d_ln_b   = (const float*)d_in[17];
    const float* d_W1     = (const float*)d_in[18];
    const float* d_b1     = (const float*)d_in[19];
    const float* d_W2     = (const float*)d_in[20];
    const float* d_b2     = (const float*)d_in[21];
    const float* out_W    = (const float*)d_in[22];
    const float* out_b    = (const float*)d_in[23];

    const size_t ACT = (size_t)MM * DD;      // 2M floats
    float* X  = (float*)d_ws;                // current activations
    float* Qb = X  + ACT;
    float* Kb = Qb + ACT;
    float* Vb = Kb + ACT;
    float* Ab = Vb + ACT;                    // attention output
    float* Tb = Ab + ACT;                    // proj / ffn2 temp
    float* Hb = Tb + ACT;                    // ffn hidden [M, FF] = 8M floats
    float* Eb = Hb + (size_t)MM * FFD;       // encoder output

    auto gemm = [&](const float* A, const float* W, const float* bias, float* C,
                    int M, int N, int K, int relu) {
        dim3 grid(N / GBN, M / GBM);
        gemm_bf16<<<grid, 256, 0, stream>>>(A, W, bias, C, M, N, K, relu);
    };
    const size_t WSZ = (size_t)DD * DD;

    // ---------------- encoder ----------------
    embed_kernel<<<(int)((ACT + 255) / 256), 256, 0, stream>>>(src, enc_emb, X, (int)ACT);
    for (int l = 0; l < LL; ++l) {
        const float* W  = e_attn_W + (size_t)l * 4 * WSZ;
        const float* bb = e_attn_b + (size_t)l * 4 * DD;
        gemm(X, W + 0 * WSZ, bb + 0 * DD, Qb, MM, DD, DD, 0);
        gemm(X, W + 1 * WSZ, bb + 1 * DD, Kb, MM, DD, DD, 0);
        gemm(X, W + 2 * WSZ, bb + 2 * DD, Vb, MM, DD, DD, 0);
        attn_kernel<<<dim3(SS / QB, HH, BB), 256, 0, stream>>>(Qb, Kb, Vb, Ab, src, 0);
        gemm(Ab, W + 3 * WSZ, bb + 3 * DD, Tb, MM, DD, DD, 0);
        add_ln_kernel<<<MM / 8, 256, 0, stream>>>(X, Tb,
            e_ln_g + (size_t)(l * 2 + 0) * DD, e_ln_b + (size_t)(l * 2 + 0) * DD, X);
        gemm(X, e_W1 + (size_t)l * DD * FFD, e_b1 + (size_t)l * FFD, Hb, MM, FFD, DD, 1);
        gemm(Hb, e_W2 + (size_t)l * FFD * DD, e_b2 + (size_t)l * DD, Tb, MM, DD, FFD, 0);
        add_ln_kernel<<<MM / 8, 256, 0, stream>>>(X, Tb,
            e_ln_g + (size_t)(l * 2 + 1) * DD, e_ln_b + (size_t)(l * 2 + 1) * DD, X);
    }
    hipMemcpyAsync(Eb, X, ACT * sizeof(float), hipMemcpyDeviceToDevice, stream);

    // ---------------- decoder ----------------
    embed_kernel<<<(int)((ACT + 255) / 256), 256, 0, stream>>>(trg, dec_emb, X, (int)ACT);
    for (int l = 0; l < LL; ++l) {
        // self-attention (pad + causal mask on trg)
        const float* Ws = d_self_W + (size_t)l * 4 * WSZ;
        const float* bs = d_self_b + (size_t)l * 4 * DD;
        gemm(X, Ws + 0 * WSZ, bs + 0 * DD, Qb, MM, DD, DD, 0);
        gemm(X, Ws + 1 * WSZ, bs + 1 * DD, Kb, MM, DD, DD, 0);
        gemm(X, Ws + 2 * WSZ, bs + 2 * DD, Vb, MM, DD, DD, 0);
        attn_kernel<<<dim3(SS / QB, HH, BB), 256, 0, stream>>>(Qb, Kb, Vb, Ab, trg, 1);
        gemm(Ab, Ws + 3 * WSZ, bs + 3 * DD, Tb, MM, DD, DD, 0);
        add_ln_kernel<<<MM / 8, 256, 0, stream>>>(X, Tb,
            d_ln_g + (size_t)(l * 3 + 0) * DD, d_ln_b + (size_t)(l * 3 + 0) * DD, X);

        // cross-attention (keys/values from encoder output, src pad mask)
        const float* Wc = d_crs_W + (size_t)l * 4 * WSZ;
        const float* bc = d_crs_b + (size_t)l * 4 * DD;
        gemm(X,  Wc + 0 * WSZ, bc + 0 * DD, Qb, MM, DD, DD, 0);
        gemm(Eb, Wc + 1 * WSZ, bc + 1 * DD, Kb, MM, DD, DD, 0);
        gemm(Eb, Wc + 2 * WSZ, bc + 2 * DD, Vb, MM, DD, DD, 0);
        attn_kernel<<<dim3(SS / QB, HH, BB), 256, 0, stream>>>(Qb, Kb, Vb, Ab, src, 0);
        gemm(Ab, Wc + 3 * WSZ, bc + 3 * DD, Tb, MM, DD, DD, 0);
        add_ln_kernel<<<MM / 8, 256, 0, stream>>>(X, Tb,
            d_ln_g + (size_t)(l * 3 + 1) * DD, d_ln_b + (size_t)(l * 3 + 1) * DD, X);

        // FFN
        gemm(X, d_W1 + (size_t)l * DD * FFD, d_b1 + (size_t)l * FFD, Hb, MM, FFD, DD, 1);
        gemm(Hb, d_W2 + (size_t)l * FFD * DD, d_b2 + (size_t)l * DD, Tb, MM, DD, FFD, 0);
        add_ln_kernel<<<MM / 8, 256, 0, stream>>>(X, Tb,
            d_ln_g + (size_t)(l * 3 + 2) * DD, d_ln_b + (size_t)(l * 3 + 2) * DD, X);
    }

    // ---------------- logits ----------------
    gemm(X, out_W, out_b, (float*)d_out, MM, VV, DD, 0);
}
